// BERT4GCN_53884659695997
// MI455X (gfx1250) — matrix operand order repaired and checked by reading, not compiled, not run on hardware
//
#include <hip/hip_runtime.h>
#include <hip/hip_bf16.h>

// ---------------------------------------------------------------------------
// Dead-code-eliminated BERT4GCN forward for MI455X (gfx1250, wave32, WMMA).
// Live computation only:
//   feature = LN(relu(gather(hs[12]) @ guid_W[3] + guid_b[3]))
//   logits  = (masked-mean_L feature) @ cls_W + cls_b
// GEMM runs as bf16 WMMA (v_wmma_f32_16x16x32_bf16) with fp32 accumulate.
// ---------------------------------------------------------------------------

#define BATCH 64
#define SEQ   256
#define DIM   768           // K of guidance GEMM (multiple of 32)
#define HID   600           // valid N
#define HP    640           // padded N (multiple of 64-wide tiles)
#define MTOT  (BATCH*SEQ)   // 16384 rows

typedef __attribute__((ext_vector_type(16))) __bf16 v16bf;
typedef __attribute__((ext_vector_type(8)))  float  v8f;

struct Q2 { uint4 a, b; };   // 32B carrier for a v16bf fragment

__device__ __forceinline__ unsigned short f2bf_raw(float f) {
    union { float f; unsigned int u; } x; x.f = f;
    unsigned int r = x.u + 0x7fffu + ((x.u >> 16) & 1u);   // round-nearest-even
    return (unsigned short)(r >> 16);
}

// ---- 1) gather hs[layer=12] rows via token_starts, convert to bf16 --------
__global__ __launch_bounds__(256)
void gather_bf16(const float* __restrict__ hs, const int* __restrict__ ts,
                 unsigned short* __restrict__ Abf) {
    // one thread = 4 contiguous floats of one row
    long long idx = (long long)blockIdx.x * blockDim.x + threadIdx.x; // MTOT*DIM/4
    if (idx >= (long long)MTOT * (DIM / 4)) return;
    int m  = (int)(idx / (DIM / 4));
    int d4 = (int)(idx - (long long)m * (DIM / 4));
    int b = m >> 8, l = m & 255;
    int t = ts[b * SEQ + l];
    const float4 v = *(const float4*)(hs + ((size_t)(12 * BATCH + b) * SEQ + t) * DIM + d4 * 4);
    unsigned short* dst = Abf + (size_t)m * DIM + d4 * 4;
    dst[0] = f2bf_raw(v.x); dst[1] = f2bf_raw(v.y);
    dst[2] = f2bf_raw(v.z); dst[3] = f2bf_raw(v.w);
}

// ---- 2) guid_W[3] (768x600 f32) -> bf16 padded to 768x640 -----------------
__global__ __launch_bounds__(256)
void wconv(const float* __restrict__ gw, unsigned short* __restrict__ Wbf) {
    int idx = blockIdx.x * blockDim.x + threadIdx.x;   // DIM*HP exact
    int k = idx / HP, n = idx - k * HP;
    float v = (n < HID) ? gw[(size_t)3 * DIM * HID + (size_t)k * HID + n] : 0.0f;
    Wbf[idx] = f2bf_raw(v);
}

// ---- 3) bf16 WMMA GEMM: C[M,HP] = A[M,DIM] @ W[DIM,HP] --------------------
// Tile 128x64 per workgroup; 8 waves in 4(M)x2(N); each wave 2x2 16x16 tiles.
__global__ __launch_bounds__(256)
void gemm_bf16wmma(const unsigned short* __restrict__ A,
                   const unsigned short* __restrict__ Bm,
                   float* __restrict__ C) {
    __shared__ __align__(16) unsigned short As[128][32];   // [m][k]   8 KB
    __shared__ __align__(16) unsigned short Bs[64][32];    // [n][k]   4 KB (transposed)

    const int tid  = threadIdx.x;
    const int lane = tid & 31;
    const int wave = tid >> 5;
    const int wm = wave & 3;          // M sub-block of 32
    const int wn = wave >> 2;         // N sub-block of 32
    const int m0 = blockIdx.x * 128;
    const int n0 = blockIdx.y * 64;
    const int half = lane >> 4;
    const int l15  = lane & 15;

    v8f acc[2][2] = {};

    for (int k0 = 0; k0 < DIM; k0 += 32) {
        // A tile: 128x32 bf16, each thread moves one 16-element half-row (32B)
        {
            int row = tid >> 1, h = tid & 1;
            const unsigned short* src = A + (size_t)(m0 + row) * DIM + k0 + h * 16;
            *(uint4*)&As[row][h * 16]     = *(const uint4*)src;
            *(uint4*)&As[row][h * 16 + 8] = *(const uint4*)(src + 8);
        }
        // B tile: 32x64 bf16, load row-major coalesced, scatter transposed
        {
            int krow = tid >> 3, cg = tid & 7;
            const unsigned short* src = Bm + (size_t)(k0 + krow) * HP + n0 + cg * 8;
            uint4 x = *(const uint4*)src;
            const unsigned short* xb = (const unsigned short*)&x;
#pragma unroll
            for (int j = 0; j < 8; ++j) Bs[cg * 8 + j][krow] = xb[j];
        }
        __syncthreads();

        // Fragments per ISA 7.12.2: A lane m=lane&15, K(e)=(e/8)*16+(lane>>4)*8+e%8
        //                           B lane n=lane&15, K(e)=(e/8)*16+(lane>>4)*8+e%8
        v16bf afr[2], bfr[2];
#pragma unroll
        for (int mi = 0; mi < 2; ++mi) {
            int mr = wm * 32 + mi * 16 + l15;
            Q2 q;
            q.a = *(const uint4*)&As[mr][half * 8];
            q.b = *(const uint4*)&As[mr][16 + half * 8];
            afr[mi] = __builtin_bit_cast(v16bf, q);
        }
#pragma unroll
        for (int ni = 0; ni < 2; ++ni) {
            int nc = wn * 32 + ni * 16 + l15;
            Q2 q;
            q.a = *(const uint4*)&Bs[nc][half * 8];
            q.b = *(const uint4*)&Bs[nc][16 + half * 8];
            bfr[ni] = __builtin_bit_cast(v16bf, q);
        }
#pragma unroll
        for (int mi = 0; mi < 2; ++mi)
#pragma unroll
            for (int ni = 0; ni < 2; ++ni)
                acc[mi][ni] = __builtin_amdgcn_wmma_f32_16x16x32_bf16(
                    false, afr[mi], false, bfr[ni], (short)0, acc[mi][ni],
                    false, false);
        __syncthreads();
    }

    // C layout: VGPR r -> M = r + 8*(lane>>4), N = lane&15
#pragma unroll
    for (int mi = 0; mi < 2; ++mi)
#pragma unroll
        for (int ni = 0; ni < 2; ++ni) {
            int gcol = n0 + wn * 32 + ni * 16 + l15;
#pragma unroll
            for (int r = 0; r < 8; ++r) {
                int grow = m0 + wm * 32 + mi * 16 + r + half * 8;
                C[(size_t)grow * HP + gcol] = acc[mi][ni][r];
            }
        }
}

// ---- 4) bias + ReLU + LayerNorm over 600 valid cols, one block per row ----
__global__ __launch_bounds__(256)
void ln_kernel(const float* __restrict__ raw, const float* __restrict__ gb,
               const float* __restrict__ lng, const float* __restrict__ lnb,
               float* __restrict__ feat) {
    const int m = blockIdx.x, tid = threadIdx.x;
    __shared__ float red[2][8];
    float x[3], s = 0.f, s2 = 0.f;
#pragma unroll
    for (int i = 0; i < 3; ++i) {
        int h = tid + i * 256;
        float v = 0.f;
        if (h < HID) v = fmaxf(raw[(size_t)m * HP + h] + gb[3 * HID + h], 0.f);
        x[i] = v; s += v; s2 += v * v;
    }
    for (int off = 16; off; off >>= 1) {
        s  += __shfl_down(s,  off, 32);
        s2 += __shfl_down(s2, off, 32);
    }
    int lane = tid & 31, w = tid >> 5;
    if (lane == 0) { red[0][w] = s; red[1][w] = s2; }
    __syncthreads();
    if (tid == 0) {
        float t = 0.f, t2 = 0.f;
        for (int i = 0; i < 8; ++i) { t += red[0][i]; t2 += red[1][i]; }
        red[0][0] = t; red[1][0] = t2;
    }
    __syncthreads();
    float mu  = red[0][0] * (1.0f / HID);
    float var = red[1][0] * (1.0f / HID) - mu * mu;
    float inv = rsqrtf(var + 1e-5f);
#pragma unroll
    for (int i = 0; i < 3; ++i) {
        int h = tid + i * 256;
        if (h < HID)
            feat[(size_t)m * HID + h] = (x[i] - mu) * inv * lng[h] + lnb[h];
    }
}

// ---- 5) masked mean over L + 600x3 classifier, one block per batch --------
__global__ __launch_bounds__(256)
void head_kernel(const float* __restrict__ feat, const float* __restrict__ amask,
                 const float* __restrict__ clsW, const float* __restrict__ clsb,
                 float* __restrict__ out) {
    const int b = blockIdx.x, tid = threadIdx.x;
    __shared__ float asp[HID];
    __shared__ float mk[SEQ];
    __shared__ float redm[8];
    __shared__ float redo[3][8];
    const int lane = tid & 31, w = tid >> 5;

    float mv = amask[b * SEQ + tid];
    mk[tid] = mv;
    float ms = mv;
    for (int off = 16; off; off >>= 1) ms += __shfl_down(ms, off, 32);
    if (lane == 0) redm[w] = ms;
    __syncthreads();
    if (tid == 0) { float t = 0.f; for (int i = 0; i < 8; ++i) t += redm[i]; redm[0] = t; }
    __syncthreads();
    const float inv_ms = 1.0f / redm[0];

    for (int h = tid; h < HID; h += 256) {
        float s = 0.f;
        for (int l = 0; l < SEQ; ++l)
            s += feat[((size_t)b * SEQ + l) * HID + h] * mk[l];
        asp[h] = s * inv_ms;
    }
    __syncthreads();

    float o0 = 0.f, o1 = 0.f, o2 = 0.f;
    for (int h = tid; h < HID; h += 256) {
        float a = asp[h];
        o0 += a * clsW[h * 3 + 0];
        o1 += a * clsW[h * 3 + 1];
        o2 += a * clsW[h * 3 + 2];
    }
    for (int off = 16; off; off >>= 1) {
        o0 += __shfl_down(o0, off, 32);
        o1 += __shfl_down(o1, off, 32);
        o2 += __shfl_down(o2, off, 32);
    }
    if (lane == 0) { redo[0][w] = o0; redo[1][w] = o1; redo[2][w] = o2; }
    __syncthreads();
    if (tid < 3) {
        float t = 0.f;
        for (int i = 0; i < 8; ++i) t += redo[tid][i];
        out[b * 3 + tid] = t + clsb[tid];
    }
}

// ---------------------------------------------------------------------------
extern "C" void kernel_launch(void* const* d_in, const int* in_sizes, int n_in,
                              void* d_out, int out_size, void* d_ws, size_t ws_size,
                              hipStream_t stream) {
    const float* hs    = (const float*)d_in[0];   // [13,64,256,768]
    const int*   ts    = (const int*)  d_in[3];   // [64,256]
    const float* amask = (const float*)d_in[5];   // [64,256]
    const float* guidW = (const float*)d_in[6];   // [4,768,600]
    const float* guidb = (const float*)d_in[7];   // [4,600]
    const float* lng   = (const float*)d_in[11];  // [600]
    const float* lnb   = (const float*)d_in[12];  // [600]
    const float* clsW  = (const float*)d_in[13];  // [600,3]
    const float* clsb  = (const float*)d_in[14];  // [3]
    float* out = (float*)d_out;                   // [64,3]

    char* ws = (char*)d_ws;
    size_t off = 0;
    unsigned short* Abf = (unsigned short*)(ws + off); off += (size_t)MTOT * DIM * 2;  // 25.2 MB
    unsigned short* Wbf = (unsigned short*)(ws + off); off += (size_t)DIM * HP * 2;    //  1.0 MB
    float*          raw = (float*)(ws + off);          off += (size_t)MTOT * HP * 4;   // 41.9 MB
    float*          feat= (float*)(ws + off);                                          // 39.3 MB

    // 1) gather layer-12 rows -> bf16
    {
        long long n = (long long)MTOT * (DIM / 4);
        int blocks = (int)((n + 255) / 256);
        gather_bf16<<<blocks, 256, 0, stream>>>(hs, ts, Abf);
    }
    // 2) weight convert (index 3)
    wconv<<<(DIM * HP) / 256, 256, 0, stream>>>(guidW, Wbf);
    // 3) WMMA GEMM
    {
        dim3 grid(MTOT / 128, HP / 64);
        gemm_bf16wmma<<<grid, 256, 0, stream>>>(Abf, Wbf, raw);
    }
    // 4) bias + relu + layernorm
    ln_kernel<<<MTOT, 256, 0, stream>>>(raw, guidb, lng, lnb, feat);
    // 5) masked mean + classifier
    head_kernel<<<BATCH, 256, 0, stream>>>(feat, amask, clsW, clsb, out);
}
